// Layer_9577777070708
// MI455X (gfx1250) — matrix operand order repaired and checked by reading, not compiled, not run on hardware
//
#include <hip/hip_runtime.h>
#include <hip/hip_bf16.h>

// ---------------------------------------------------------------------------
// Problem dims (compile-time, from reference)
// ---------------------------------------------------------------------------
#define D_Hc   1024
#define D_FFc  4096
#define BSc    8
#define Pc     2048
#define Mrows  (BSc * Pc)          // 16384 rows in every GEMM

typedef __attribute__((ext_vector_type(16))) __bf16 v16bf;
typedef __attribute__((ext_vector_type(8)))  float  v8f;
typedef int v4i_vs __attribute__((vector_size(16)));   // matches builtin's V4i

#if defined(__HIP_DEVICE_COMPILE__) && \
    __has_builtin(__builtin_amdgcn_global_load_async_to_lds_b128)
#define HAVE_ASYNC_LDS 1
#else
#define HAVE_ASYNC_LDS 0
#endif

enum { MODE_AB = 0, MODE_WO = 1, MODE_FFN1 = 2, MODE_FFN2 = 3 };

struct GemmArgs {
  const float*  x0;        // MODE_AB: x_all   (K-segment 0)
  const float*  x1;        // MODE_AB: y_pm_all
  const float*  x2;        // MODE_AB: y_wm_proj_all
  const float*  x3;        // MODE_AB: y_em_proj_all
  const float*  Af32;      // f32 A operand (WO / FFN1)
  const __bf16* Abf16;     // bf16 A operand (FFN2)
  const __bf16* Wt;        // B operand, pre-transposed bf16  [N][K]
  int           K;         // reduction length (multiple of 64)
  int           ldC;       // output row stride
  const float*  bias;      // b_ab / b_o / b1 / b2
  const float*  surprise;  // [BS]
  const float*  wab_last;  // W_ab row 4096 (f32, length 2048)
  const int*    carry;     // [M]
  const float*  res;       // residual source (x_all for WO, out for FFN2)
  float*        aeff;      // MODE_AB outputs
  float*        bbuf;
  float*        Cf32;      // f32 output
  __bf16*       Cbf16;     // bf16 output (FFN1 -> FFN2 A operand)
};

// ---------------------------------------------------------------------------
// bf16 WMMA GEMM: block tile 128x128, K-tile 64, double-buffered LDS,
// 8 waves, wave tile 64x32, async global->LDS for the B (weight) tiles.
// ---------------------------------------------------------------------------
#define TM  128
#define TN  128
#define TK  64
#define LDP 72   // LDS row stride in bf16 elems (144B, bank-conflict pad)

// ---- A-operand global fetch into registers (f32 modes convert later) ------
template<int MODE>
__device__ __forceinline__ void gemm_load_a(const GemmArgs& g, int m0, int srow,
                                            int kglob, float4 f[8], uint4 q[4]) {
  if constexpr (MODE == MODE_FFN2) {
    const uint4* sp = (const uint4*)(g.Abf16 + (size_t)(m0 + srow) * g.K + kglob);
#pragma unroll
    for (int i = 0; i < 4; ++i) q[i] = sp[i];
  } else {
    const float* ap;
    if constexpr (MODE == MODE_AB) {
      const int ksel = kglob >> 10;          // 32-elem chunk stays in one seg
      const float* base = (ksel == 0) ? g.x0
                        : (ksel == 1) ? g.x1
                        : (ksel == 2) ? g.x2 : g.x3;
      ap = base + (size_t)(m0 + srow) * D_Hc + (kglob & (D_Hc - 1));
    } else {
      ap = g.Af32 + (size_t)(m0 + srow) * g.K + kglob;
    }
    const float4* ap4 = (const float4*)ap;
#pragma unroll
    for (int i = 0; i < 8; ++i) f[i] = ap4[i];
  }
}

// ---- A-operand store regs -> LDS (with f32->bf16 conversion) ---------------
template<int MODE>
__device__ __forceinline__ void gemm_store_a(__bf16* dst, const float4 f[8],
                                             const uint4 q[4]) {
  if constexpr (MODE == MODE_FFN2) {
#pragma unroll
    for (int i = 0; i < 4; ++i) *(uint4*)(dst + 8 * i) = q[i];
  } else {
    union { __bf16 h[32]; uint4 v[4]; } u;
#pragma unroll
    for (int i = 0; i < 8; ++i) {
      u.h[4 * i + 0] = (__bf16)f[i].x;
      u.h[4 * i + 1] = (__bf16)f[i].y;
      u.h[4 * i + 2] = (__bf16)f[i].z;
      u.h[4 * i + 3] = (__bf16)f[i].w;
    }
#pragma unroll
    for (int i = 0; i < 4; ++i) *(uint4*)(dst + 8 * i) = u.v[i];
  }
}

// ---- B-operand: straight bf16 copy global -> LDS (async on CDNA5) ----------
__device__ __forceinline__ void gemm_stage_b(const __bf16* gsrc, __bf16* ldst) {
#if HAVE_ASYNC_LDS
#pragma unroll
  for (int i = 0; i < 4; ++i) {
    __builtin_amdgcn_global_load_async_to_lds_b128(
        (v4i_vs*)(gsrc + 8 * i),     // generic -> __device__ AS (implicit)
        (v4i_vs*)(ldst + 8 * i),     // generic -> __shared__ AS (implicit)
        0, 0);
  }
#else
  const uint4* bp = (const uint4*)gsrc;
#pragma unroll
  for (int i = 0; i < 4; ++i) *(uint4*)(ldst + 8 * i) = bp[i];
#endif
}

__device__ __forceinline__ void gemm_wait_async() {
#if HAVE_ASYNC_LDS
#if __has_builtin(__builtin_amdgcn_s_wait_asynccnt)
  __builtin_amdgcn_s_wait_asynccnt(0);
#else
  asm volatile("s_wait_asynccnt 0" ::: "memory");
#endif
#endif
}

template<int MODE>
__global__ __launch_bounds__(256)
void gemm_bf16(GemmArgs g) {
  __shared__ __bf16 lA[2][TM][LDP];
  __shared__ __bf16 lB[2][TN][LDP];

  const int tid   = threadIdx.x;
  const int m0    = blockIdx.y * TM;
  const int n0    = blockIdx.x * TN;
  const int lane  = tid & 31;
  const int wid   = tid >> 5;
  const int waveM = (wid >> 2) * 64;   // 0 / 64
  const int waveN = (wid & 3)  * 32;   // 0..96
  const int lrow  = lane & 15;
  const int lkg   = lane >> 4;

  // staging role: thread -> (row 0..127, 32-elem k-chunk 0/32)
  const int srow = tid >> 1;
  const int skc  = (tid & 1) * 32;

  v8f acc[4][2] = {};

  const int ntiles = g.K / TK;

  // ---------------- prologue: stage tile 0 into buffer 0 -------------------
  {
    float4 f[8]; uint4 q[4];
    gemm_load_a<MODE>(g, m0, srow, skc, f, q);
    gemm_stage_b(g.Wt + (size_t)(n0 + srow) * g.K + skc, &lB[0][srow][skc]);
    gemm_store_a<MODE>(&lA[0][srow][skc], f, q);
    gemm_wait_async();
  }
  __syncthreads();

  for (int kt = 0; kt < ntiles; ++kt) {
    const int cur = kt & 1;
    const int nxt = cur ^ 1;
    const bool hasNext = (kt + 1) < ntiles;

    // ---- prefetch next tile: A -> registers, B -> async LDS (buffer nxt) --
    float4 fpre[8]; uint4 qpre[4];
    if (hasNext) {
      const int kg = (kt + 1) * TK + skc;
      gemm_load_a<MODE>(g, m0, srow, kg, fpre, qpre);
      gemm_stage_b(g.Wt + (size_t)(n0 + srow) * g.K + kg, &lB[nxt][srow][skc]);
    }

    // ---- compute on current buffer: 2 substeps x 8 WMMAs ------------------
#pragma unroll
    for (int ks = 0; ks < TK; ks += 32) {
      v16bf afr[4], bfr[2];
#pragma unroll
      for (int i = 0; i < 4; ++i) {
        const __bf16* r = &lA[cur][waveM + i * 16 + lrow][ks];
        union { v16bf v; uint4 q[2]; } u;
        u.q[0] = *(const uint4*)(r + lkg * 8);        // K = kg*8 .. +7
        u.q[1] = *(const uint4*)(r + 16 + lkg * 8);   // K = 16+kg*8 .. +7
        afr[i] = u.v;
      }
#pragma unroll
      for (int j = 0; j < 2; ++j) {
        const __bf16* r = &lB[cur][waveN + j * 16 + lrow][ks];
        union { v16bf v; uint4 q[2]; } u;
        u.q[0] = *(const uint4*)(r + lkg * 8);        // K = kg*8 .. +7
        u.q[1] = *(const uint4*)(r + 16 + lkg * 8);   // K = 16+kg*8 .. +7
        bfr[j] = u.v;
      }
#pragma unroll
      for (int i = 0; i < 4; ++i)
#pragma unroll
        for (int j = 0; j < 2; ++j)
          acc[i][j] = __builtin_amdgcn_wmma_f32_16x16x32_bf16(
              false, afr[i], false, bfr[j], (short)0, acc[i][j], false, false);
    }

    // ---- finish staging next tile, single barrier per K-tile --------------
    if (hasNext) {
      gemm_store_a<MODE>(&lA[nxt][srow][skc], fpre, qpre);
      gemm_wait_async();
    }
    __syncthreads();
  }

  // ------------------- fused epilogues ----------------------------------
#pragma unroll
  for (int i = 0; i < 4; ++i) {
#pragma unroll
    for (int j = 0; j < 2; ++j) {
#pragma unroll
      for (int v = 0; v < 8; ++v) {
        const int gm = m0 + waveM + i * 16 + lkg * 8 + v;
        const int gn = n0 + waveN + j * 16 + lrow;
        float val = acc[i][j][v];
        if constexpr (MODE == MODE_AB) {
          const int bs = gm >> 11;                       // row -> batch
          val += g.bias[gn] + g.surprise[bs] * g.wab_last[gn];
          if (gn < D_Hc) {
            const float s = 1.0f / (1.0f + __expf(-val));
            g.aeff[(size_t)gm * D_Hc + gn] = s * (float)g.carry[gm];
          } else {
            g.bbuf[(size_t)gm * D_Hc + (gn - D_Hc)] = tanhf(val);
          }
        } else if constexpr (MODE == MODE_WO) {
          val += g.bias[gn] + g.res[(size_t)gm * D_Hc + gn];
          g.Cf32[(size_t)gm * g.ldC + gn] = val;
        } else if constexpr (MODE == MODE_FFN1) {
          val += g.bias[gn];
          const float t =
              tanhf(0.7978845608028654f * (val + 0.044715f * val * val * val));
          g.Cbf16[(size_t)gm * g.ldC + gn] = (__bf16)(0.5f * val * (1.0f + t));
        } else {  // MODE_FFN2: final output = out + ffn_out
          val += g.bias[gn] + g.res[(size_t)gm * D_Hc + gn];
          g.Cf32[(size_t)gm * g.ldC + gn] = val;
        }
      }
    }
  }
}

// ---------------------------------------------------------------------------
// Weight transpose + f32 -> bf16:  in [K][N] f32  ->  out [N][K] bf16
// ---------------------------------------------------------------------------
__global__ __launch_bounds__(256)
void transpose_cvt(const float* __restrict__ in, __bf16* __restrict__ out,
                   int K, int N) {
  __shared__ float tile[32][33];
  const int tx = threadIdx.x, ty = threadIdx.y;
  const int gn0 = blockIdx.x * 32, gk0 = blockIdx.y * 32;
  for (int r = ty; r < 32; r += 8)
    tile[r][tx] = in[(size_t)(gk0 + r) * N + gn0 + tx];
  __syncthreads();
  for (int r = ty; r < 32; r += 8)
    out[(size_t)(gn0 + r) * K + gk0 + tx] = (__bf16)tile[tx][r];
}

// ---------------------------------------------------------------------------
// Chunked affine scan  h_t = a_t*h_{t-1} + b_t   (P=2048 = 32 chunks x 64)
// ---------------------------------------------------------------------------
#define NCH 32
#define CLEN 64

__global__ __launch_bounds__(256)
void scan_compose(const float* __restrict__ a, const float* __restrict__ b,
                  float* __restrict__ cA, float* __restrict__ cB) {
  const int tid = blockIdx.x * 256 + threadIdx.x;  // BS*DH*NCH = 262144
  const int d  = tid & (D_Hc - 1);
  const int r  = tid >> 10;           // bs*NCH + c
  const int c  = r & (NCH - 1);
  const int bs = r >> 5;
  size_t base = ((size_t)(bs * Pc + c * CLEN)) * D_Hc + d;
  float A = 1.0f, B = 0.0f;
  for (int t = 0; t < CLEN; ++t) {
    const float at = a[base + (size_t)t * D_Hc];
    const float bt = b[base + (size_t)t * D_Hc];
    A = at * A;
    B = at * B + bt;
  }
  cA[(size_t)r * D_Hc + d] = A;
  cB[(size_t)r * D_Hc + d] = B;
}

__global__ __launch_bounds__(256)
void scan_prefix(const float* __restrict__ cA, const float* __restrict__ cB,
                 const float* __restrict__ h0, float* __restrict__ pref) {
  const int tid = blockIdx.x * 256 + threadIdx.x;  // BS*DH = 8192
  const int d = tid & (D_Hc - 1);
  const int bs = tid >> 10;
  float h = h0[(size_t)bs * D_Hc + d];
  for (int c = 0; c < NCH; ++c) {
    const size_t idx = (size_t)(bs * NCH + c) * D_Hc + d;
    pref[idx] = h;
    h = cA[idx] * h + cB[idx];
  }
}

__global__ __launch_bounds__(256)
void scan_apply(const float* __restrict__ a, const float* __restrict__ b,
                const float* __restrict__ pref, float* __restrict__ h_all) {
  const int tid = blockIdx.x * 256 + threadIdx.x;
  const int d  = tid & (D_Hc - 1);
  const int r  = tid >> 10;
  const int c  = r & (NCH - 1);
  const int bs = r >> 5;
  size_t base = ((size_t)(bs * Pc + c * CLEN)) * D_Hc + d;
  float h = pref[(size_t)r * D_Hc + d];
  for (int t = 0; t < CLEN; ++t) {
    const size_t idx = base + (size_t)t * D_Hc;
    h = a[idx] * h + b[idx];
    h_all[idx] = h;
  }
}

// ---------------------------------------------------------------------------
// Fused double layernorm: out = LN(z), ffn_in = LN(out).  One block per row.
// ---------------------------------------------------------------------------
__global__ __launch_bounds__(256)
void double_ln(const float* __restrict__ z,
               const float* __restrict__ g1, const float* __restrict__ b1,
               const float* __restrict__ g2, const float* __restrict__ b2,
               float* __restrict__ outb, float* __restrict__ ffin) {
  __shared__ float s1[256], s2[256];
  const int row = blockIdx.x, t = threadIdx.x;
  const float* zr = z + (size_t)row * D_Hc;
  float v[4], sum = 0.f, sq = 0.f;
#pragma unroll
  for (int j = 0; j < 4; ++j) {
    v[j] = zr[t + 256 * j];
    sum += v[j]; sq += v[j] * v[j];
  }
  s1[t] = sum; s2[t] = sq; __syncthreads();
  for (int o = 128; o > 0; o >>= 1) {
    if (t < o) { s1[t] += s1[t + o]; s2[t] += s2[t + o]; }
    __syncthreads();
  }
  const float mu = s1[0] * (1.0f / D_Hc);
  const float rs = rsqrtf(s2[0] * (1.0f / D_Hc) - mu * mu + 1e-5f);
  float ov[4]; sum = 0.f; sq = 0.f;
#pragma unroll
  for (int j = 0; j < 4; ++j) {
    const int n = t + 256 * j;
    ov[j] = (v[j] - mu) * rs * g1[n] + b1[n];
    sum += ov[j]; sq += ov[j] * ov[j];
  }
  __syncthreads();
  s1[t] = sum; s2[t] = sq; __syncthreads();
  for (int o = 128; o > 0; o >>= 1) {
    if (t < o) { s1[t] += s1[t + o]; s2[t] += s2[t + o]; }
    __syncthreads();
  }
  const float mu2 = s1[0] * (1.0f / D_Hc);
  const float rs2 = rsqrtf(s2[0] * (1.0f / D_Hc) - mu2 * mu2 + 1e-5f);
#pragma unroll
  for (int j = 0; j < 4; ++j) {
    const int n = t + 256 * j;
    outb[(size_t)row * D_Hc + n] = ov[j];
    ffin[(size_t)row * D_Hc + n] = (ov[j] - mu2) * rs2 * g2[n] + b2[n];
  }
}

// ---------------------------------------------------------------------------
// Host-side orchestration
// ---------------------------------------------------------------------------
extern "C" void kernel_launch(void* const* d_in, const int* in_sizes, int n_in,
                              void* d_out, int out_size, void* d_ws,
                              size_t ws_size, hipStream_t stream) {
  (void)in_sizes; (void)n_in; (void)out_size; (void)ws_size;
  const float* x_all      = (const float*)d_in[0];
  const float* y_pm       = (const float*)d_in[1];
  const float* y_wm       = (const float*)d_in[2];
  const float* y_em       = (const float*)d_in[3];
  const float* surprise   = (const float*)d_in[4];
  const float* h0         = (const float*)d_in[5];
  const float* W_ab       = (const float*)d_in[6];
  const float* b_ab       = (const float*)d_in[7];
  const float* W_o        = (const float*)d_in[8];
  const float* b_o        = (const float*)d_in[9];
  const float* norm_g     = (const float*)d_in[10];
  const float* norm_b     = (const float*)d_in[11];
  const float* ffn_g      = (const float*)d_in[12];
  const float* ffn_b      = (const float*)d_in[13];
  const float* W1         = (const float*)d_in[14];
  const float* b1         = (const float*)d_in[15];
  const float* W2         = (const float*)d_in[16];
  const float* b2         = (const float*)d_in[17];
  const int*   carry      = (const int*)d_in[18];

  char* ws = (char*)d_ws;
  size_t off = 0;
  auto alloc = [&](size_t bytes) -> void* {
    void* p = ws + off;
    off = (off + bytes + 255) & ~(size_t)255;
    return p;
  };
  __bf16* Wt_ab = (__bf16*)alloc((size_t)2048 * 4096 * 2);
  __bf16* Wt_o  = (__bf16*)alloc((size_t)1024 * 1024 * 2);
  __bf16* Wt_1  = (__bf16*)alloc((size_t)4096 * 1024 * 2);
  __bf16* Wt_2  = (__bf16*)alloc((size_t)1024 * 4096 * 2);
  float*  aeff  = (float*)alloc((size_t)Mrows * D_Hc * 4);   // later: z
  float*  bbuf  = (float*)alloc((size_t)Mrows * D_Hc * 4);   // later: ffn_in
  float*  hall  = (float*)alloc((size_t)Mrows * D_Hc * 4);   // later: out
  __bf16* gbuf  = (__bf16*)alloc((size_t)Mrows * D_FFc * 2);
  float*  cA    = (float*)alloc((size_t)BSc * NCH * D_Hc * 4);
  float*  cB    = (float*)alloc((size_t)BSc * NCH * D_Hc * 4);
  float*  pref  = (float*)alloc((size_t)BSc * NCH * D_Hc * 4);

  // 1) weights -> bf16, N-major
  dim3 tb(32, 8);
  transpose_cvt<<<dim3(2048 / 32, 4096 / 32), tb, 0, stream>>>(W_ab, Wt_ab, 4096, 2048);
  transpose_cvt<<<dim3(1024 / 32, 1024 / 32), tb, 0, stream>>>(W_o,  Wt_o,  1024, 1024);
  transpose_cvt<<<dim3(4096 / 32, 1024 / 32), tb, 0, stream>>>(W1,   Wt_1,  1024, 4096);
  transpose_cvt<<<dim3(1024 / 32, 4096 / 32), tb, 0, stream>>>(W2,   Wt_2,  4096, 1024);

  // 2) GEMM1: ab = u @ W_ab (+bias, +surprise rank-1) -> sigmoid/tanh/carry
  {
    GemmArgs a = {};
    a.x0 = x_all; a.x1 = y_pm; a.x2 = y_wm; a.x3 = y_em;
    a.Wt = Wt_ab; a.K = 4096; a.ldC = 2048;
    a.bias = b_ab; a.surprise = surprise;
    a.wab_last = W_ab + (size_t)4096 * 2048;
    a.carry = carry; a.aeff = aeff; a.bbuf = bbuf;
    gemm_bf16<MODE_AB><<<dim3(2048 / TN, Mrows / TM), 256, 0, stream>>>(a);
  }

  // 3) affine scan along P
  scan_compose<<<(BSc * D_Hc * NCH) / 256, 256, 0, stream>>>(aeff, bbuf, cA, cB);
  scan_prefix<<<(BSc * D_Hc) / 256, 256, 0, stream>>>(cA, cB, h0, pref);
  scan_apply<<<(BSc * D_Hc * NCH) / 256, 256, 0, stream>>>(aeff, bbuf, pref, hall);

  // 4) z = h @ W_o + b_o + x   (write into aeff, now free)
  float* zbuf = aeff;
  {
    GemmArgs a = {};
    a.Af32 = hall; a.Wt = Wt_o; a.K = 1024; a.ldC = 1024;
    a.bias = b_o; a.res = x_all; a.Cf32 = zbuf;
    gemm_bf16<MODE_WO><<<dim3(1024 / TN, Mrows / TM), 256, 0, stream>>>(a);
  }

  // 5) out = LN(z); ffn_in = LN(out)   (out -> hall, ffn_in -> bbuf)
  float* outb = hall;
  float* ffin = bbuf;
  double_ln<<<Mrows, 256, 0, stream>>>(zbuf, norm_g, norm_b, ffn_g, ffn_b,
                                       outb, ffin);

  // 6) g = gelu(ffn_in @ W1 + b1)  (bf16, feeds FFN2 A operand directly)
  {
    GemmArgs a = {};
    a.Af32 = ffin; a.Wt = Wt_1; a.K = 1024; a.ldC = D_FFc;
    a.bias = b1; a.Cbf16 = gbuf;
    gemm_bf16<MODE_FFN1><<<dim3(D_FFc / TN, Mrows / TM), 256, 0, stream>>>(a);
  }

  // 7) result = out + g @ W2 + b2
  {
    GemmArgs a = {};
    a.Abf16 = gbuf; a.Wt = Wt_2; a.K = 4096; a.ldC = 1024;
    a.bias = b2; a.res = outb; a.Cf32 = (float*)d_out;
    gemm_bf16<MODE_FFN2><<<dim3(1024 / TN, Mrows / TM), 256, 0, stream>>>(a);
  }
}
